// VariableSelectionNetwork_62414464745975
// MI455X (gfx1250) — matrix operand order repaired and checked by reading, not compiled, not run on hardware
//
#include <hip/hip_runtime.h>

// ---------------------------------------------------------------------------
// Types for CDNA5 WMMA (wave32, V_WMMA_F32_16X16X32_BF16)
// ---------------------------------------------------------------------------
typedef __bf16 bf16;
typedef __attribute__((ext_vector_type(16))) __bf16 v16bf;
typedef __attribute__((ext_vector_type(8)))  __bf16 v8bf;
typedef __attribute__((ext_vector_type(8)))  float  v8f;

// Problem constants
enum : int { kB = 32, kT = 256, kV = 16, kH = 256, kVH = 4096, kTok = kB * kT, kM = 16 };

// Packed bf16 weight workspace offsets (in bf16 elements)
enum : size_t {
  OJWA    = 0,                         // 4096x256
  OJWC    = OJWA    + 4096u * 256u,    // 256x256
  OJWI    = OJWC    + 256u  * 256u,    // 256x256
  OJWGLU  = OJWI    + 256u  * 256u,    // 256x32
  OJWSKIP = OJWGLU  + 256u  * 32u,     // 4096x16
  OVWA    = OJWSKIP + 4096u * 16u,     // 16 x 256x256
  OVWI    = OVWA    + 16u * 256u * 256u,
  OVWGLU  = OVWI    + 16u * 256u * 256u, // 16 x 256x512
  OWS_END = OVWGLU  + 16u * 256u * 512u
};

static __device__ inline float bf2f(bf16 b)  { return static_cast<float>(b); }
static __device__ inline bf16  f2bf(float f) { return static_cast<bf16>(f); }
static __device__ inline float eluf(float x)  { return x > 0.f ? x : __expf(x) - 1.f; }
static __device__ inline float sigmf(float x) { return 1.f / (1.f + __expf(-x)); }

static __device__ inline v8f wmma_bf16(v16bf a, v16bf b, v8f c) {
  // D = A(16x32 bf16) * B(32x16 bf16) + C(16x16 f32)
  return __builtin_amdgcn_wmma_f32_16x16x32_bf16(false, a, false, b, (short)0, c, false, false);
}

// A-fragment (16x32 bf16) from an LDS bf16 buffer, row-major [16][stride].
// Lane l: row m = l&15; lanes<16 cover K 0..7 & 16..23, lanes>=16 cover 8..15 & 24..31.
static __device__ inline v16bf afrag_lds(const bf16* buf, int stride, int kk, int lane) {
  int m   = lane & 15;
  int hof = (lane < 16) ? 0 : 8;
  const v8bf* p = reinterpret_cast<const v8bf*>(buf + (size_t)m * stride + kk * 32 + hof);
  v8bf lo = p[0];   // K base..base+7
  v8bf hi = p[2];   // K base+16..base+23
  return __builtin_shufflevector(lo, hi, 0,1,2,3,4,5,6,7,8,9,10,11,12,13,14,15);
}

// B-fragment from pre-packed weights: one contiguous 32B chunk per lane.
static __device__ inline v16bf bfrag(const bf16* __restrict__ pk, int frag, int lane) {
  return *reinterpret_cast<const v16bf*>(pk + ((size_t)frag << 9) + (lane << 4));
}

// Store a WMMA C fragment (f32 16x16) as bf16 into LDS row-major [16][256].
// C layout: lane l, vgpr r -> (m = r + (l<16?0:8), n = l&15)
static __device__ inline void store_c_lds(bf16* buf, int ntile, int lane, v8f acc) {
  int n  = ntile * 16 + (lane & 15);
  int mb = (lane < 16) ? 0 : 8;
  v8bf cb = __builtin_convertvector(acc, v8bf);
#pragma unroll
  for (int r = 0; r < 8; ++r) buf[(mb + r) * kH + n] = cb[r];
}

// ---------------------------------------------------------------------------
// Kernel 0: pack f32 weights into bf16 WMMA-B fragment order.
// Packed element idx = mat*K*N + frag*512 + lane*16 + e,
//   frag = kstep*(N/16)+ntile ; n = ntile*16+(lane&15) ;
//   k = kstep*32 + (e<8 ? (lane<16?0:8) + e : 16 + (lane<16?0:8) + e-8)
// ---------------------------------------------------------------------------
__global__ void vsn_pack_kernel(const float* __restrict__ src, bf16* __restrict__ dst,
                                int K, int N, int nmat) {
  size_t total = (size_t)nmat * K * N;
  size_t idx = (size_t)blockIdx.x * blockDim.x + threadIdx.x;
  if (idx >= total) return;
  size_t matsz = (size_t)K * N;
  size_t mat = idx / matsz;
  size_t w   = idx - mat * matsz;
  int e    = (int)(w & 15);
  int lane = (int)((w >> 4) & 31);
  size_t frag = w >> 9;
  int ntiles = N >> 4;
  int kstep = (int)(frag / ntiles);
  int ntile = (int)(frag % ntiles);
  int n = ntile * 16 + (lane & 15);
  int k = kstep * 32 + ((lane < 16) ? 0 : 8) + (e & 7) + ((e >> 3) << 4);
  dst[idx] = f2bf(src[mat * matsz + (size_t)k * N + n]);
}

// ---------------------------------------------------------------------------
// Main fused VSN kernel: one block per 16-token tile, 256 threads (8 waves).
// x tile staged ONCE in LDS as bf16; all WMMA A-fragments come from LDS.
// ---------------------------------------------------------------------------
__global__ __launch_bounds__(256) void vsn_main_kernel(
    const float* __restrict__ x,        // [8192, 16, 256] (flat [tok][4096])
    const float* __restrict__ ctx,      // [8192, 256]
    const float* __restrict__ jba,      // [256]
    const float* __restrict__ jbi,      // [256]
    const float* __restrict__ jbglu,    // [32]
    const float* __restrict__ jbskip,   // [16]
    const float* __restrict__ jlng,     // [16]
    const float* __restrict__ jlnb,     // [16]
    const float* __restrict__ vba,      // [16,256]
    const float* __restrict__ vbi,      // [16,256]
    const float* __restrict__ vbglu,    // [16,512]
    const float* __restrict__ vlng,     // [16,256]
    const float* __restrict__ vlnb,     // [16,256]
    const bf16*  __restrict__ wpk,      // packed weights
    float* __restrict__ out)            // [8192*256] ctx-out ++ [8192*16] weights
{
  __shared__ __align__(32) bf16  s_x[kM * kVH];     // 128 KB: x tile, bf16
  __shared__ __align__(32) bf16  s_ctx[kM * kH];    // context tile, bf16
  __shared__ __align__(32) bf16  s_a[kM * kH];      // elu(joint pre-act)
  __shared__ __align__(32) bf16  s_i[kM * kH];      // joint hidden
  __shared__ float s_g[kM * 32];                    // GLU logits
  __shared__ float s_skip[kM * kV];                 // skip projection (atomic-reduced)
  __shared__ float s_sw[kM * kV];                   // softmax weights
  __shared__ float s_out[kM * kH];                  // weighted output accumulator
  __shared__ __align__(32) bf16  s_av[4][kM * kH];  // per-pair staging (av, reused for h)
  __shared__ __align__(32) bf16  s_iv[4][kM * kH];  // per-pair staging (iv)

  const int tid  = threadIdx.x;
  const int lane = tid & 31;
  const int wave = tid >> 5;
  const int tok0 = blockIdx.x * kM;
  const int nlo  = lane & 15;
  const int mb   = (lane < 16) ? 0 : 8;

  // ---- stage 0: init accumulators; stage x tile + context tile as bf16 ----
#pragma unroll
  for (int j = 0; j < 16; ++j) s_out[tid * 16 + j] = 0.f;
  s_skip[tid] = 0.f;
  // x tile: 16 x 4096 = 65536 elems, 8-elem chunks, 32 chunks per thread
#pragma unroll 4
  for (int j = 0; j < 32; ++j) {
    int flat = (j * 256 + tid) * 8;
    int m = flat >> 12, col = flat & 4095;
    const float4* q = reinterpret_cast<const float4*>(x + (size_t)(tok0 + m) * kVH + col);
    float4 a = q[0], b = q[1];
    v8f f = {a.x, a.y, a.z, a.w, b.x, b.y, b.z, b.w};
    *reinterpret_cast<v8bf*>(s_x + flat) = __builtin_convertvector(f, v8bf);
  }
  // context tile: 16 x 256 = 4096 elems, 2 chunks per thread
#pragma unroll
  for (int j = 0; j < 2; ++j) {
    int flat = (j * 256 + tid) * 8;
    int m = flat >> 8, col = flat & 255;
    const float4* q = reinterpret_cast<const float4*>(ctx + (size_t)(tok0 + m) * kH + col);
    float4 a = q[0], b = q[1];
    v8f f = {a.x, a.y, a.z, a.w, b.x, b.y, b.z, b.w};
    *reinterpret_cast<v8bf*>(s_ctx + flat) = __builtin_convertvector(f, v8bf);
  }
  __syncthreads();

  // ---- stage 1: a = elu(Xi @ jWa + ba + ctx @ jWc) ; each wave owns 2 N-tiles ----
  {
    const int ntA = wave * 2, ntB = ntA + 1;
    v8f acc0 = {0,0,0,0,0,0,0,0}, acc1 = {0,0,0,0,0,0,0,0};
    const bf16* WA = wpk + OJWA;
#pragma unroll 4
    for (int kk = 0; kk < 128; ++kk) {
      v16bf a = afrag_lds(s_x, kVH, kk, lane);
      // speculative prefetch: safe even past the matrix end (stays inside ws)
      __builtin_prefetch(WA + (((size_t)(kk + 8) * 16 + ntA) << 9) + (lane << 4), 0, 0);
      acc0 = wmma_bf16(a, bfrag(WA, kk * 16 + ntA, lane), acc0);
      acc1 = wmma_bf16(a, bfrag(WA, kk * 16 + ntB, lane), acc1);
    }
    const bf16* WC = wpk + OJWC;
#pragma unroll
    for (int kk = 0; kk < 8; ++kk) {
      v16bf a = afrag_lds(s_ctx, kH, kk, lane);
      acc0 = wmma_bf16(a, bfrag(WC, kk * 16 + ntA, lane), acc0);
      acc1 = wmma_bf16(a, bfrag(WC, kk * 16 + ntB, lane), acc1);
    }
    float b0 = jba[ntA * 16 + nlo], b1 = jba[ntB * 16 + nlo];
#pragma unroll
    for (int r = 0; r < 8; ++r) { acc0[r] = eluf(acc0[r] + b0); acc1[r] = eluf(acc1[r] + b1); }
    store_c_lds(s_a, ntA, lane, acc0);
    store_c_lds(s_a, ntB, lane, acc1);
  }
  __syncthreads();

  // ---- stage 2: i = a @ jWi + bi ----
  {
    const int ntA = wave * 2, ntB = ntA + 1;
    v8f acc0 = {0,0,0,0,0,0,0,0}, acc1 = {0,0,0,0,0,0,0,0};
    const bf16* WI = wpk + OJWI;
#pragma unroll
    for (int kk = 0; kk < 8; ++kk) {
      v16bf a = afrag_lds(s_a, kH, kk, lane);
      acc0 = wmma_bf16(a, bfrag(WI, kk * 16 + ntA, lane), acc0);
      acc1 = wmma_bf16(a, bfrag(WI, kk * 16 + ntB, lane), acc1);
    }
    float b0 = jbi[ntA * 16 + nlo], b1 = jbi[ntB * 16 + nlo];
#pragma unroll
    for (int r = 0; r < 8; ++r) { acc0[r] += b0; acc1[r] += b1; }
    store_c_lds(s_i, ntA, lane, acc0);
    store_c_lds(s_i, ntB, lane, acc1);
  }
  __syncthreads();

  // ---- stage 3: g = i @ jWglu (waves 0-1) ; skip = Xi @ jWskip (waves 2-5, K-split) ----
  if (wave < 2) {
    const bf16* WG = wpk + OJWGLU;     // 256x32 -> 2 n-tiles
    v8f acc = {0,0,0,0,0,0,0,0};
#pragma unroll
    for (int kk = 0; kk < 8; ++kk)
      acc = wmma_bf16(afrag_lds(s_i, kH, kk, lane), bfrag(WG, kk * 2 + wave, lane), acc);
    float bb = jbglu[wave * 16 + nlo];
#pragma unroll
    for (int r = 0; r < 8; ++r) s_g[(mb + r) * 32 + wave * 16 + nlo] = acc[r] + bb;
  } else if (wave < 6) {
    const bf16* WS = wpk + OJWSKIP;    // 4096x16 -> 1 n-tile, 128 k-steps
    const int q = wave - 2;
    v8f acc = {0,0,0,0,0,0,0,0};
#pragma unroll 4
    for (int j = 0; j < 32; ++j) {
      int kk = q * 32 + j;
      acc = wmma_bf16(afrag_lds(s_x, kVH, kk, lane), bfrag(WS, kk, lane), acc);
    }
#pragma unroll
    for (int r = 0; r < 8; ++r) atomicAdd(&s_skip[(mb + r) * kV + nlo], acc[r]);
  }
  __syncthreads();

  // ---- stage 4: GLU + skip + LayerNorm(V) + softmax -> sparse weights ----
  if (tid < kM) {
    const int m = tid;
    float val[kV];
    float mean = 0.f;
#pragma unroll
    for (int vv = 0; vv < kV; ++vv) {
      float g1 = s_g[m * 32 + vv], g2 = s_g[m * 32 + 16 + vv];
      float t = g1 * sigmf(g2) + s_skip[m * kV + vv] + jbskip[vv];
      val[vv] = t; mean += t;
    }
    mean *= (1.f / kV);
    float var = 0.f;
#pragma unroll
    for (int vv = 0; vv < kV; ++vv) { float d = val[vv] - mean; var += d * d; }
    var *= (1.f / kV);
    float rstd = __frsqrt_rn(var + 1e-5f);
    float mx = -3.4e38f;
#pragma unroll
    for (int vv = 0; vv < kV; ++vv) {
      val[vv] = (val[vv] - mean) * rstd * jlng[vv] + jlnb[vv];
      mx = fmaxf(mx, val[vv]);
    }
    float sum = 0.f;
#pragma unroll
    for (int vv = 0; vv < kV; ++vv) { val[vv] = __expf(val[vv] - mx); sum += val[vv]; }
    float inv = 1.f / sum;
    float* swout = out + (size_t)kTok * kH;
#pragma unroll
    for (int vv = 0; vv < kV; ++vv) {
      float w = val[vv] * inv;
      s_sw[m * kV + vv] = w;
      swout[(size_t)(tok0 + m) * kV + vv] = w;
    }
  }
  __syncthreads();

  // ---- stage 5: per-variable GRNs. 4 wave-pairs, each pair owns one variable per
  //      iteration; sub-wave handles 8 N-tiles. ----
  const int pair = wave >> 1;
  const int sub  = wave & 1;
  for (int it = 0; it < 4; ++it) {
    const int v = it * 4 + pair;
    bf16* bav = s_av[pair];
    bf16* biv = s_iv[pair];

    // 5a: av = elu(x_v @ vWa[v] + vba[v])
    {
      v16bf A[8];
#pragma unroll
      for (int kk = 0; kk < 8; ++kk) A[kk] = afrag_lds(s_x + v * kH, kVH, kk, lane);
      const bf16* W = wpk + OVWA + (size_t)v * (256u * 256u);
      for (int nt8 = 0; nt8 < 8; ++nt8) {
        int nt = sub * 8 + nt8;
        v8f acc = {0,0,0,0,0,0,0,0};
#pragma unroll
        for (int kk = 0; kk < 8; ++kk) acc = wmma_bf16(A[kk], bfrag(W, kk * 16 + nt, lane), acc);
        float bb = vba[v * kH + nt * 16 + nlo];
#pragma unroll
        for (int r = 0; r < 8; ++r) acc[r] = eluf(acc[r] + bb);
        store_c_lds(bav, nt, lane, acc);
      }
    }
    __syncthreads();

    // 5b: iv = av @ vWi[v] + vbi[v]
    {
      v16bf A[8];
#pragma unroll
      for (int kk = 0; kk < 8; ++kk) A[kk] = afrag_lds(bav, kH, kk, lane);
      const bf16* W = wpk + OVWI + (size_t)v * (256u * 256u);
      for (int nt8 = 0; nt8 < 8; ++nt8) {
        int nt = sub * 8 + nt8;
        v8f acc = {0,0,0,0,0,0,0,0};
#pragma unroll
        for (int kk = 0; kk < 8; ++kk) acc = wmma_bf16(A[kk], bfrag(W, kk * 16 + nt, lane), acc);
        float bb = vbi[v * kH + nt * 16 + nlo];
#pragma unroll
        for (int r = 0; r < 8; ++r) acc[r] += bb;
        store_c_lds(biv, nt, lane, acc);
      }
    }
    __syncthreads();

    // 5c: gv = iv @ vWglu[v] + vbglu[v]; h = gv1*sigmoid(gv2) + x_v  (h -> reuse bav)
    {
      v16bf A[8];
#pragma unroll
      for (int kk = 0; kk < 8; ++kk) A[kk] = afrag_lds(biv, kH, kk, lane);
      const bf16* W = wpk + OVWGLU + (size_t)v * (256u * 512u);   // 256x512: 32 n-tiles
      for (int nt8 = 0; nt8 < 8; ++nt8) {
        int nt = sub * 8 + nt8;
        v8f acc1 = {0,0,0,0,0,0,0,0}, acc2 = {0,0,0,0,0,0,0,0};
#pragma unroll
        for (int kk = 0; kk < 8; ++kk) {
          acc1 = wmma_bf16(A[kk], bfrag(W, kk * 32 + nt,      lane), acc1);
          acc2 = wmma_bf16(A[kk], bfrag(W, kk * 32 + nt + 16, lane), acc2);
        }
        float b1 = vbglu[v * 512 + nt * 16 + nlo];
        float b2 = vbglu[v * 512 + 256 + nt * 16 + nlo];
#pragma unroll
        for (int r = 0; r < 8; ++r) {
          int m = mb + r;
          float xr = bf2f(s_x[(size_t)m * kVH + v * kH + nt * 16 + nlo]);
          float h = (acc1[r] + b1) * sigmf(acc2[r] + b2) + xr;
          bav[m * kH + nt * 16 + nlo] = f2bf(h);
        }
      }
    }
    __syncthreads();

    // 5d: per-token LayerNorm(H) of h, scale by sparse weight, accumulate output.
    // All 64 lanes of the pair active: 4 lanes per token, 64 columns per lane,
    // vectorized LDS loads, cross-lane reduction via lane permutes.
    {
      const int m  = sub * 8 + (lane >> 2);   // token handled by this lane
      const int c0 = (lane & 3) * 64;         // 64-column slice
      const v8bf* hp = reinterpret_cast<const v8bf*>(bav + m * kH + c0);
      float sum = 0.f, ss = 0.f;
#pragma unroll
      for (int j = 0; j < 8; ++j) {
        v8f f = __builtin_convertvector(hp[j], v8f);
#pragma unroll
        for (int e = 0; e < 8; ++e) { sum += f[e]; ss += f[e] * f[e]; }
      }
      // reduce across the 4 lanes of this token (lane-permute hardware)
      sum += __shfl_xor(sum, 1, 32);  ss += __shfl_xor(ss, 1, 32);
      sum += __shfl_xor(sum, 2, 32);  ss += __shfl_xor(ss, 2, 32);
      float mean = sum * (1.f / kH);
      float var  = ss * (1.f / kH) - mean * mean;
      float rstd = __frsqrt_rn(var + 1e-5f);
      float w = s_sw[m * kV + v];
      const float4* gg = reinterpret_cast<const float4*>(vlng + v * kH + c0);
      const float4* bb = reinterpret_cast<const float4*>(vlnb + v * kH + c0);
#pragma unroll
      for (int j = 0; j < 8; ++j) {
        v8f f = __builtin_convertvector(hp[j], v8f);
        float4 g0 = gg[j * 2], g1 = gg[j * 2 + 1];
        float4 b0 = bb[j * 2], b1 = bb[j * 2 + 1];
        float gv[8] = {g0.x, g0.y, g0.z, g0.w, g1.x, g1.y, g1.z, g1.w};
        float bv[8] = {b0.x, b0.y, b0.z, b0.w, b1.x, b1.y, b1.z, b1.w};
#pragma unroll
        for (int e = 0; e < 8; ++e) {
          int c = c0 + j * 8 + e;
          float t = (f[e] - mean) * rstd * gv[e] + bv[e];
          atomicAdd(&s_out[m * kH + c], t * w);
        }
      }
    }
    __syncthreads();
  }

  // ---- stage 6: write variable_ctx tile ----
#pragma unroll
  for (int j = 0; j < 16; ++j) {
    int idx = tid * 16 + j;
    int m = idx >> 8, c = idx & 255;
    out[(size_t)(tok0 + m) * kH + c] = s_out[idx];
  }
}

// ---------------------------------------------------------------------------
// Host launcher
// ---------------------------------------------------------------------------
extern "C" void kernel_launch(void* const* d_in, const int* in_sizes, int n_in,
                              void* d_out, int out_size, void* d_ws, size_t ws_size,
                              hipStream_t stream) {
  (void)in_sizes; (void)n_in; (void)out_size; (void)ws_size;
  const float* x       = (const float*)d_in[0];
  const float* ctx     = (const float*)d_in[1];
  const float* jWa     = (const float*)d_in[2];
  const float* jba     = (const float*)d_in[3];
  const float* jWc     = (const float*)d_in[4];
  const float* jWi     = (const float*)d_in[5];
  const float* jbi     = (const float*)d_in[6];
  const float* jWglu   = (const float*)d_in[7];
  const float* jbglu   = (const float*)d_in[8];
  const float* jWskip  = (const float*)d_in[9];
  const float* jbskip  = (const float*)d_in[10];
  const float* jlng    = (const float*)d_in[11];
  const float* jlnb    = (const float*)d_in[12];
  const float* vWa     = (const float*)d_in[13];
  const float* vba     = (const float*)d_in[14];
  const float* vWi     = (const float*)d_in[15];
  const float* vbi     = (const float*)d_in[16];
  const float* vWglu   = (const float*)d_in[17];
  const float* vbglu   = (const float*)d_in[18];
  const float* vlng    = (const float*)d_in[19];
  const float* vlnb    = (const float*)d_in[20];
  float* out = (float*)d_out;
  bf16* ws   = (bf16*)d_ws;

  auto pack = [&](const float* src, size_t off, int K, int N, int nmat) {
    size_t total = (size_t)nmat * K * N;
    int blocks = (int)((total + 255) / 256);
    vsn_pack_kernel<<<blocks, 256, 0, stream>>>(src, ws + off, K, N, nmat);
  };
  pack(jWa,    OJWA,    4096, 256, 1);
  pack(jWc,    OJWC,    256,  256, 1);
  pack(jWi,    OJWI,    256,  256, 1);
  pack(jWglu,  OJWGLU,  256,  32,  1);
  pack(jWskip, OJWSKIP, 4096, 16,  1);
  pack(vWa,    OVWA,    256,  256, 16);
  pack(vWi,    OVWI,    256,  256, 16);
  pack(vWglu,  OVWGLU,  256,  512, 16);

  vsn_main_kernel<<<kTok / kM, 256, 0, stream>>>(
      x, ctx, jba, jbi, jbglu, jbskip, jlng, jlnb,
      vba, vbi, vbglu, vlng, vlnb, ws, out);
}
